// CrystalGNN_84155589198110
// MI455X (gfx1250) — compile-verified, matrix operand order checked by reading, not run on hardware
//
#include <hip/hip_runtime.h>
#include <hip/hip_bf16.h>

// ---------------- problem constants ----------------
#define NN   50000          // nodes
#define EE   800000         // edges
#define DIM  128            // feature dim (D == H)
#define EDIM 50             // edge-attr dim
#define NG   256            // graphs
#define NET  (EE + NN)      // edges + self loops

typedef __attribute__((ext_vector_type(16))) __bf16 v16bf;
typedef __attribute__((ext_vector_type(8)))  float  v8f;

__device__ __forceinline__ unsigned short f2bf(float f) {
    unsigned int u = __float_as_uint(f);
    unsigned int r = u + 0x7FFFu + ((u >> 16) & 1u);   // round-to-nearest-even
    return (unsigned short)(r >> 16);
}

// ---------------- prep kernels ----------------

// column sums of edge_attr [E x 50] -> colsum[50]
__global__ __launch_bounds__(256) void edge_colsum_kernel(
    const float* __restrict__ ea, float* __restrict__ colsum) {
    __shared__ float sm[EDIM];
    if (threadIdx.x < EDIM) sm[threadIdx.x] = 0.f;
    __syncthreads();
    int e = blockIdx.x * 256 + threadIdx.x;
    if (e < EE) {
        const float* row = ea + (size_t)e * EDIM;
        #pragma unroll 10
        for (int k = 0; k < EDIM; ++k) atomicAdd(&sm[k], row[k]);
    }
    __syncthreads();
    if (threadIdx.x < EDIM) unsafeAtomicAdd(&colsum[threadIdx.x], sm[threadIdx.x]);
}

// we{1,2}[k] = We{1,2}[k,:] . ae{1,2};  scal[l] = (colsum/E) . we{l}
__global__ void prep_small_kernel(
    const float* __restrict__ We1, const float* __restrict__ ae1,
    const float* __restrict__ We2, const float* __restrict__ ae2,
    const float* __restrict__ colsum,
    float* __restrict__ we1, float* __restrict__ we2, float* __restrict__ scal) {
    int k = threadIdx.x;
    if (k < EDIM) {
        float s1 = 0.f, s2 = 0.f;
        for (int j = 0; j < DIM; ++j) {
            s1 += We1[k * DIM + j] * ae1[j];
            s2 += We2[k * DIM + j] * ae2[j];
        }
        we1[k] = s1; we2[k] = s2;
    }
    __syncthreads();
    if (k == 0) {
        const float invE = 1.0f / (float)EE;
        float l1 = 0.f, l2 = 0.f;
        for (int j = 0; j < EDIM; ++j) {
            float m = colsum[j] * invE;
            l1 += m * we1[j]; l2 += m * we2[j];
        }
        scal[0] = l1; scal[1] = l2;
    }
}

// bf16 transpose-pack of W1/W2: Wt[n*128+k] = bf16(W[k*128+n])
__global__ void pack_w_kernel(const float* __restrict__ W1, const float* __restrict__ W2,
                              unsigned short* __restrict__ Wt1, unsigned short* __restrict__ Wt2) {
    int i = blockIdx.x * 256 + threadIdx.x;
    if (i >= DIM * DIM) return;
    int n = i >> 7, k = i & 127;
    Wt1[n * DIM + k] = f2bf(W1[k * DIM + n]);
    Wt2[n * DIM + k] = f2bf(W2[k * DIM + n]);
}

__global__ __launch_bounds__(256) void cvt_bf16_kernel(
    const float* __restrict__ src, unsigned short* __restrict__ dst, int n) {
    int i = blockIdx.x * 256 + threadIdx.x;
    if (i < n) dst[i] = f2bf(src[i]);
}

// per-edge edge-attn scalars for both layers
__global__ __launch_bounds__(256) void edge_ale_kernel(
    const float* __restrict__ ea, const float* __restrict__ we1,
    const float* __restrict__ we2, float* __restrict__ ale1, float* __restrict__ ale2) {
    int e = blockIdx.x * 256 + threadIdx.x;
    if (e >= EE) return;
    const float* row = ea + (size_t)e * EDIM;
    float s1 = 0.f, s2 = 0.f;
    #pragma unroll 10
    for (int k = 0; k < EDIM; ++k) { float v = row[k]; s1 += v * we1[k]; s2 += v * we2[k]; }
    ale1[e] = s1; ale2[e] = s2;
}

// ---------------- WMMA GEMM: Y[M x 128] = Xbf[M x 128] @ W (Wt = bf16 col-major) ----------
// Tiles staged global->LDS via CDNA5 GLOBAL_LOAD_ASYNC_TO_LDS_B128 (ASYNCcnt),
// compute via v_wmma_f32_16x16x32_bf16, one 16x16x128 tile per wave.
__global__ __launch_bounds__(256) void gemm_bf16_wmma_kernel(
    const unsigned short* __restrict__ Xbf,   // [M x 128] bf16 row-major
    const unsigned short* __restrict__ WtBf,  // [128 x 128] bf16, Wt[n][k] = W[k][n]
    float* __restrict__ Y) {
    __shared__ unsigned short Xs[16 * 128];    // 4 KB
    __shared__ unsigned short Ws[128 * 128];   // 32 KB
    const int tid = threadIdx.x;
    const int rowbase = blockIdx.x * 16;

    // ---- async global -> LDS tile staging (no VGPR round-trip) ----
    {
        // X tile: 256 lanes x 16B = 4 KB in one async op
        const unsigned ldsX = (unsigned)(size_t)Xs + (unsigned)tid * 16u;
        const unsigned offX = (unsigned)tid * 16u;
        const unsigned long long gaX =
            (unsigned long long)(Xbf + (size_t)rowbase * DIM);
        asm volatile("global_load_async_to_lds_b128 %0, %1, %2"
                     :: "v"(ldsX), "v"(offX), "s"(gaX) : "memory");
        // W tile: 8 x (256 lanes x 16B) = 32 KB
        const unsigned ldsW0 = (unsigned)(size_t)Ws;
        const unsigned long long gaW = (unsigned long long)WtBf;
        #pragma unroll
        for (int j = 0; j < 8; ++j) {
            const unsigned off = (unsigned)tid * 16u + (unsigned)j * 4096u;
            const unsigned ldsW = ldsW0 + off;
            asm volatile("global_load_async_to_lds_b128 %0, %1, %2"
                         :: "v"(ldsW), "v"(off), "s"(gaW) : "memory");
        }
        // drain this wave's ASYNCcnt before signaling the workgroup barrier
        asm volatile("s_wait_asynccnt 0" ::: "memory");
    }
    __syncthreads();

    const int wave = tid >> 5;          // 8 waves -> 8 col tiles of 16
    const int lane = tid & 31;
    const int half = lane >> 4;
    const int rc   = lane & 15;         // A row / B,D column within tile
    const int cb   = wave << 4;

    union Frag { v16bf v; unsigned int u[8]; };
    v8f acc = {};
    #pragma unroll
    for (int k0 = 0; k0 < 128; k0 += 32) {
        Frag a, b;
        #pragma unroll
        for (int j = 0; j < 8; ++j) {
            // K indices for packed pair j per CDNA5 16-bit A layout (lanes split K halves)
            const int kk = k0 + ((j & 3) << 1) + (half << 3) + ((j >> 2) << 4);
            a.u[j] = *(const unsigned int*)&Xs[rc * DIM + kk];
            b.u[j] = *(const unsigned int*)&Ws[(cb + rc) * DIM + kk];
        }
        acc = __builtin_amdgcn_wmma_f32_16x16x32_bf16(false, a.v, false, b.v,
                                                      (short)0, acc, false, false);
    }
    // D layout: VGPR v -> row = v + 8*half, col = lane&15
    float* out = Y + (size_t)(rowbase + half * 8) * DIM + cb + rc;
    #pragma unroll
    for (int v = 0; v < 8; ++v) out[(size_t)v * DIM] = acc[v];
}

// ---------------- attention kernels ----------------

__global__ __launch_bounds__(256) void node_alpha_kernel(
    const float* __restrict__ h, const float* __restrict__ a_s,
    const float* __restrict__ a_d, float* __restrict__ als, float* __restrict__ ald) {
    int i = blockIdx.x * 256 + threadIdx.x;
    if (i >= NN) return;
    const float4* hr = (const float4*)(h + (size_t)i * DIM);
    const float4* s4 = (const float4*)a_s;
    const float4* d4 = (const float4*)a_d;
    float ss = 0.f, dd = 0.f;
    #pragma unroll
    for (int k = 0; k < DIM / 4; ++k) {
        float4 hv = hr[k], sv = s4[k], dv = d4[k];
        ss += hv.x * sv.x + hv.y * sv.y + hv.z * sv.z + hv.w * sv.w;
        dd += hv.x * dv.x + hv.y * dv.y + hv.z * dv.z + hv.w * dv.w;
    }
    als[i] = ss; ald[i] = dd;
}

// pass 1: alpha = leaky_relu(als[src]+ald[dst]+ale), atomic-max per dst (ordered uint keys)
__global__ __launch_bounds__(256) void p1_alpha_max_kernel(
    const int* __restrict__ ei, const float* __restrict__ ale,
    const float* __restrict__ scal, int layer,
    const float* __restrict__ als, const float* __restrict__ ald,
    float* __restrict__ alpha, unsigned int* __restrict__ amaxk) {
    int i = blockIdx.x * 256 + threadIdx.x;
    if (i >= NET) return;
    int s, d; float ae;
    if (i < EE) { s = ei[i]; d = ei[EE + i]; ae = ale[i]; }
    else        { s = d = i - EE; ae = scal[layer]; }
    float a = als[s] + ald[d] + ae;
    a = a > 0.f ? a : 0.2f * a;                 // leaky relu, slope 0.2
    alpha[i] = a;
    unsigned int u = __float_as_uint(a);
    unsigned int key = (u & 0x80000000u) ? ~u : (u | 0x80000000u);
    atomicMax(&amaxk[d], key);
}

// pass 2: ex = exp(alpha - amax[dst]); denom[dst] += ex
__global__ __launch_bounds__(256) void p2_exp_sum_kernel(
    const int* __restrict__ ei, float* __restrict__ alpha,
    const unsigned int* __restrict__ amaxk, float* __restrict__ denom) {
    int i = blockIdx.x * 256 + threadIdx.x;
    if (i >= NET) return;
    int d = (i < EE) ? ei[EE + i] : (i - EE);
    unsigned int key = amaxk[d];
    unsigned int u = (key & 0x80000000u) ? (key & 0x7FFFFFFFu) : ~key;
    float amax = __uint_as_float(u);
    float ex = __expf(alpha[i] - amax);
    alpha[i] = ex;
    unsafeAtomicAdd(&denom[d], ex);
}

// pass 3: out[dst,:] += h[src,:] * (ex / denom[dst]) -- one wave32 per edge
__global__ __launch_bounds__(256) void p3_scatter_kernel(
    const int* __restrict__ ei, const float* __restrict__ alpha,
    const float* __restrict__ denom, const float* __restrict__ h,
    float* __restrict__ out) {
    int w = (blockIdx.x * 256 + threadIdx.x) >> 5;
    int lane = threadIdx.x & 31;
    if (w >= NET) return;
    int s, d;
    if (w < EE) { s = ei[w]; d = ei[EE + w]; } else { s = d = w - EE; }
    float coef = alpha[w] / denom[d];
    float4 v = ((const float4*)(h + (size_t)s * DIM))[lane];
    float* o = out + (size_t)d * DIM + lane * 4;
    __builtin_prefetch(o, 1, 3);                // global_prefetch of dst row
    unsafeAtomicAdd(o + 0, v.x * coef);
    unsafeAtomicAdd(o + 1, v.y * coef);
    unsafeAtomicAdd(o + 2, v.z * coef);
    unsafeAtomicAdd(o + 3, v.w * coef);
}

__global__ __launch_bounds__(256) void bias_relu_kernel(
    float* __restrict__ o, const float* __restrict__ b) {
    int i = blockIdx.x * 256 + threadIdx.x;
    if (i >= NN * DIM) return;
    float v = o[i] + b[i & (DIM - 1)];
    o[i] = v > 0.f ? v : 0.f;
}

// ---------------- pooling + final linear ----------------
__global__ __launch_bounds__(256) void pool_kernel(
    const float* __restrict__ h, const int* __restrict__ batch,
    float* __restrict__ sums, float* __restrict__ cnt) {
    int w = (blockIdx.x * 256 + threadIdx.x) >> 5;
    int lane = threadIdx.x & 31;
    if (w >= NN) return;
    int g = batch[w];
    float4 v = ((const float4*)(h + (size_t)w * DIM))[lane];
    float* s = sums + (size_t)g * DIM + lane * 4;
    unsafeAtomicAdd(s + 0, v.x);
    unsafeAtomicAdd(s + 1, v.y);
    unsafeAtomicAdd(s + 2, v.z);
    unsafeAtomicAdd(s + 3, v.w);
    if (lane == 0) unsafeAtomicAdd(&cnt[g], 1.0f);
}

__global__ __launch_bounds__(128) void final_kernel(
    const float* __restrict__ sums, const float* __restrict__ cnt,
    const float* __restrict__ Wfc, const float* __restrict__ bfc,
    float* __restrict__ out) {
    int g = blockIdx.x, t = threadIdx.x;
    float c = cnt[g]; c = c > 1.f ? c : 1.f;
    float v = (sums[(size_t)g * DIM + t] / c) * Wfc[t];
    #pragma unroll
    for (int off = 16; off; off >>= 1) v += __shfl_down(v, off, 32);
    __shared__ float sm[4];
    if ((t & 31) == 0) sm[t >> 5] = v;
    __syncthreads();
    if (t == 0) out[g] = sm[0] + sm[1] + sm[2] + sm[3] + bfc[0];
}

// ---------------- host launcher ----------------
extern "C" void kernel_launch(void* const* d_in, const int* in_sizes, int n_in,
                              void* d_out, int out_size, void* d_ws, size_t ws_size,
                              hipStream_t stream) {
    const float* x    = (const float*)d_in[0];
    const int*   ei   = (const int*)  d_in[1];
    const float* ea   = (const float*)d_in[2];
    const int*   bat  = (const int*)  d_in[3];
    const float* W1   = (const float*)d_in[4];
    const float* as1  = (const float*)d_in[5];
    const float* ad1  = (const float*)d_in[6];
    const float* We1  = (const float*)d_in[7];
    const float* ae1  = (const float*)d_in[8];
    const float* b1   = (const float*)d_in[9];
    const float* W2   = (const float*)d_in[10];
    const float* as2  = (const float*)d_in[11];
    const float* ad2  = (const float*)d_in[12];
    const float* We2  = (const float*)d_in[13];
    const float* ae2  = (const float*)d_in[14];
    const float* b2   = (const float*)d_in[15];
    const float* Wfc  = (const float*)d_in[16];
    const float* bfc  = (const float*)d_in[17];
    float* out = (float*)d_out;

    float* ws = (float*)d_ws;
    // workspace layout (float units)
    const size_t NH = (size_t)NN * DIM;
    size_t oA    = 0;                 // g1 / g2 buffer (GEMM out)     N*H
    size_t oB    = oA + NH;           // o1 / o2 buffer (aggregate)    N*H
    size_t oXbf  = oB + NH;           // bf16 staging (N*H u16)        N*H/2
    size_t oWt1  = oXbf + NH / 2;     // bf16 W1^T                     8192
    size_t oWt2  = oWt1 + 8192;       // bf16 W2^T                     8192
    size_t oCol  = oWt2 + 8192;       // colsum                        64
    size_t oWe1  = oCol + 64;         // we1                           64
    size_t oWe2  = oWe1 + 64;         // we2                           64
    size_t oScal = oWe2 + 64;         // self-loop scalars             64
    size_t oAle1 = oScal + 64;        // per-edge attn scalar L1       E
    size_t oAle2 = oAle1 + EE;        // per-edge attn scalar L2       E
    size_t oAls  = oAle2 + EE;        // node src alpha                N
    size_t oAld  = oAls + NN;         // node dst alpha                N
    size_t oAlp  = oAld + NN;         // per-edge alpha/ex             E+N
    size_t oAmx  = oAlp + NET;        // max keys (uint)               N
    size_t oDen  = oAmx + NN;         // softmax denom                 N
    size_t oSum  = oDen + NN;         // pooled sums                   G*H
    size_t oCnt  = oSum + (size_t)NG * DIM;   // counts                G

    float* bufA = ws + oA;
    float* bufB = ws + oB;
    unsigned short* xbf = (unsigned short*)(ws + oXbf);
    unsigned short* wt1 = (unsigned short*)(ws + oWt1);
    unsigned short* wt2 = (unsigned short*)(ws + oWt2);
    float* colsum = ws + oCol;
    float* we1 = ws + oWe1; float* we2 = ws + oWe2; float* scal = ws + oScal;
    float* ale1 = ws + oAle1; float* ale2 = ws + oAle2;
    float* als = ws + oAls; float* ald = ws + oAld;
    float* alpha = ws + oAlp;
    unsigned int* amaxk = (unsigned int*)(ws + oAmx);
    float* denom = ws + oDen;
    float* sums = ws + oSum; float* cnt = ws + oCnt;

    const int EB = (EE + 255) / 256;           // 3125
    const int NB = (NN + 255) / 256;           // 196
    const int NEB = (NET + 255) / 256;         // 3321
    const int WEB = (NET + 7) / 8;             // wave-per-edge blocks
    const int ELB = (int)(NH + 255) / 256;     // 25000
    const int GB = NN / 16;                    // 3125 GEMM tiles

    // ---- prep ----
    hipMemsetAsync(colsum, 0, 64 * sizeof(float), stream);
    hipMemsetAsync(bufB, 0, NH * sizeof(float), stream);
    hipMemsetAsync(amaxk, 0, 2 * NN * sizeof(float), stream);   // amax keys + denom
    edge_colsum_kernel<<<EB, 256, 0, stream>>>(ea, colsum);
    prep_small_kernel<<<1, 64, 0, stream>>>(We1, ae1, We2, ae2, colsum, we1, we2, scal);
    pack_w_kernel<<<64, 256, 0, stream>>>(W1, W2, wt1, wt2);
    cvt_bf16_kernel<<<ELB, 256, 0, stream>>>(x, xbf, (int)NH);
    edge_ale_kernel<<<EB, 256, 0, stream>>>(ea, we1, we2, ale1, ale2);

    // ---- layer 1 ----
    gemm_bf16_wmma_kernel<<<GB, 256, 0, stream>>>(xbf, wt1, bufA);
    node_alpha_kernel<<<NB, 256, 0, stream>>>(bufA, as1, ad1, als, ald);
    p1_alpha_max_kernel<<<NEB, 256, 0, stream>>>(ei, ale1, scal, 0, als, ald, alpha, amaxk);
    p2_exp_sum_kernel<<<NEB, 256, 0, stream>>>(ei, alpha, amaxk, denom);
    p3_scatter_kernel<<<WEB, 256, 0, stream>>>(ei, alpha, denom, bufA, bufB);
    bias_relu_kernel<<<ELB, 256, 0, stream>>>(bufB, b1);

    // ---- layer 2 ----
    cvt_bf16_kernel<<<ELB, 256, 0, stream>>>(bufB, xbf, (int)NH);
    hipMemsetAsync(amaxk, 0, 2 * NN * sizeof(float), stream);   // re-zero keys + denom
    gemm_bf16_wmma_kernel<<<GB, 256, 0, stream>>>(xbf, wt2, bufA);
    node_alpha_kernel<<<NB, 256, 0, stream>>>(bufA, as2, ad2, als, ald);
    p1_alpha_max_kernel<<<NEB, 256, 0, stream>>>(ei, ale2, scal, 1, als, ald, alpha, amaxk);
    p2_exp_sum_kernel<<<NEB, 256, 0, stream>>>(ei, alpha, amaxk, denom);
    hipMemsetAsync(bufB, 0, NH * sizeof(float), stream);        // o1 dead after cvt
    p3_scatter_kernel<<<WEB, 256, 0, stream>>>(ei, alpha, denom, bufA, bufB);
    bias_relu_kernel<<<ELB, 256, 0, stream>>>(bufB, b2);

    // ---- pool + fc ----
    hipMemsetAsync(sums, 0, ((size_t)NG * DIM + NG) * sizeof(float), stream);
    pool_kernel<<<(NN + 7) / 8, 256, 0, stream>>>(bufB, bat, sums, cnt);
    final_kernel<<<NG, 128, 0, stream>>>(sums, cnt, Wfc, bfc, out);
}
